// MoE_82592221102591
// MI455X (gfx1250) — compile-verified, matrix operand order checked by reading, not run on hardware
//
#include <hip/hip_runtime.h>
#include <hip/hip_bf16.h>

#define HIDDEN   2048
#define N_EXP    8
#define GDIM     64
#define BTOK     4096
#define NOISE_STD 0.01f
#define BAL_COEF  0.01f

typedef __attribute__((ext_vector_type(16))) __bf16        v16bf;
typedef __attribute__((ext_vector_type(8)))  float         v8f;
typedef __attribute__((ext_vector_type(4)))  unsigned int  v4u;
typedef __attribute__((ext_vector_type(2)))  unsigned int  v2u;

union FragBF { v16bf v; v4u q[2]; };

__device__ __forceinline__ unsigned short f2bf(float f) {
    unsigned int u = __float_as_uint(f);
    u += 0x7FFFu + ((u >> 16) & 1u);          // round-to-nearest-even
    return (unsigned short)(u >> 16);
}

__device__ __forceinline__ unsigned int pack_bf2(float lo, float hi) {
    return ((unsigned int)f2bf(hi) << 16) | (unsigned int)f2bf(lo);
}

__device__ __forceinline__ v8f zero8() {
    v8f z;
#pragma unroll
    for (int i = 0; i < 8; ++i) z[i] = 0.0f;
    return z;
}

__device__ __forceinline__ float sigmoidf(float x) {
    return 1.0f / (1.0f + __expf(-x));
}

// ---------------------------------------------------------------- init
__global__ void moe_init_kernel(float* gateSum) {
    if (threadIdx.x < N_EXP) gateSum[threadIdx.x] = 0.0f;
}

// ---------------------------------------------------------------- gating
__global__ void moe_gating_kernel(const float* __restrict__ x,
                                  const float* __restrict__ noise,
                                  const float* __restrict__ gw1,
                                  const float* __restrict__ gb1,
                                  const float* __restrict__ gw2,
                                  const float* __restrict__ gb2,
                                  float* __restrict__ comb,
                                  float* __restrict__ combsum,
                                  float* __restrict__ gateSum) {
    __shared__ float sx[HIDDEN];
    __shared__ float sh[GDIM];
    __shared__ float sg[N_EXP];

    const int row = blockIdx.x;
    const int t   = threadIdx.x;

    for (int i = t; i < HIDDEN; i += 64)
        sx[i] = x[(size_t)row * HIDDEN + i];
    __syncthreads();

    float acc = gb1[t];
    for (int k = 0; k < HIDDEN; ++k)
        acc = fmaf(sx[k], gw1[(size_t)k * GDIM + t], acc);
    sh[t] = fmaxf(acc, 0.0f);
    __syncthreads();

    if (t < N_EXP) {
        float a = gb2[t];
#pragma unroll
        for (int j = 0; j < GDIM; ++j)
            a = fmaf(sh[j], gw2[j * N_EXP + t], a);
        sg[t] = a;
    }
    __syncthreads();

    if (t == 0) {
        float g[N_EXP];
        float m = sg[0];
#pragma unroll
        for (int e = 1; e < N_EXP; ++e) m = fmaxf(m, sg[e]);
        float s = 0.0f;
#pragma unroll
        for (int e = 0; e < N_EXP; ++e) { g[e] = __expf(sg[e] - m); s += g[e]; }
        float inv = 1.0f / s;
#pragma unroll
        for (int e = 0; e < N_EXP; ++e)
            g[e] = g[e] * inv + noise[(size_t)row * N_EXP + e] * NOISE_STD;

#pragma unroll
        for (int e = 0; e < N_EXP; ++e)
            atomicAdd(&gateSum[e], g[e]);

        int i1 = 0;
#pragma unroll
        for (int e = 1; e < N_EXP; ++e) if (g[e] > g[i1]) i1 = e;
        int i2 = (i1 == 0) ? 1 : 0;
#pragma unroll
        for (int e = 0; e < N_EXP; ++e)
            if (e != i1 && g[e] > g[i2]) i2 = e;

        float w1 = g[i1], w2 = g[i2];
        float isum = 1.0f / (w1 + w2);
        w1 *= isum; w2 *= isum;

#pragma unroll
        for (int e = 0; e < N_EXP; ++e) {
            float c = 0.0f;
            if (e == i1) c = w1;
            if (e == i2) c = w2;
            comb[(size_t)row * N_EXP + e] = c;
        }
        combsum[row] = w1 + w2;
    }
}

// ---------------------------------------------------------------- pre-pass: x -> bf16 (row-major [m][k])
__global__ void moe_cvt_x_kernel(const float* __restrict__ x,
                                 unsigned short* __restrict__ xbf) {
    size_t base = ((size_t)blockIdx.x * 256 + threadIdx.x) * 8;
    float4 a = *(const float4*)&x[base];
    float4 b = *(const float4*)&x[base + 4];
    v4u p;
    p.x = pack_bf2(a.x, a.y);
    p.y = pack_bf2(a.z, a.w);
    p.z = pack_bf2(b.x, b.y);
    p.w = pack_bf2(b.z, b.w);
    *(v4u*)&xbf[base] = p;
}

// ---------------------------------------------------------------- pre-pass: W -> bf16 transposed [n][k]
// grid = (2048/64, 2048/64, 9); matrix 0 = shared_w, 1..8 = experts
__global__ void moe_cvt_w_kernel(const float* __restrict__ shared_w,
                                 const float* __restrict__ expert_w,
                                 unsigned short* __restrict__ wt) {
    __shared__ alignas(16) float sT[64 * 68];

    const int n0  = blockIdx.x * 64;
    const int k0  = blockIdx.y * 64;
    const int mat = blockIdx.z;
    const int t   = threadIdx.x;

    const float* src = (mat == 0) ? shared_w
                                  : expert_w + (size_t)(mat - 1) * HIDDEN * HIDDEN;
    unsigned short* dst = wt + (size_t)mat * HIDDEN * HIDDEN;

#pragma unroll
    for (int i = 0; i < 4; ++i) {                 // load 64x64 f32 tile
        int idx = t + i * 256;
        int r  = idx >> 4;
        int c4 = (idx & 15) << 2;
        *(float4*)&sT[r * 68 + c4] =
            *(const float4*)&src[(size_t)(k0 + r) * HIDDEN + n0 + c4];
    }
    __syncthreads();
#pragma unroll
    for (int i = 0; i < 4; ++i) {                 // write transposed bf16
        int idx = t + i * 256;
        int n  = idx >> 4;
        int k4 = (idx & 15) << 2;
        float f0 = sT[(k4 + 0) * 68 + n];
        float f1 = sT[(k4 + 1) * 68 + n];
        float f2 = sT[(k4 + 2) * 68 + n];
        float f3 = sT[(k4 + 3) * 68 + n];
        v2u p; p.x = pack_bf2(f0, f1); p.y = pack_bf2(f2, f3);
        *(v2u*)&dst[(size_t)(n0 + n) * HIDDEN + k0 + k4] = p;
    }
}

// ---------------------------------------------------------------- fused GEMM
// Block tile 128x128, 8 waves in 2(M)x4(N); each wave: 4x2 16x16 tiles = 8 wmma/K-step.
#define BM 128
#define BN 128
#define BK 32
#define LDS_STRIDE 40   // elements; 80B row stride, 16B aligned, conflict-free

template <bool PRE>
__global__ __launch_bounds__(256)
void moe_gemm_kernel(const float* __restrict__ x,
                     const unsigned short* __restrict__ xbf,
                     const unsigned short* __restrict__ wt,
                     const float* __restrict__ shared_w,
                     const float* __restrict__ shared_b,
                     const float* __restrict__ shared_scale,
                     const float* __restrict__ expert_w,
                     const float* __restrict__ expert_b,
                     const float* __restrict__ router_scale,
                     const float* __restrict__ comb,
                     const float* __restrict__ combsum,
                     float* __restrict__ out) {
    __shared__ alignas(16) unsigned short sA[BM * LDS_STRIDE]; // [m][k] bf16
    __shared__ alignas(16) unsigned short sB[BN * LDS_STRIDE]; // [n][k] bf16
    __shared__ float sComb[BM * N_EXP];
    __shared__ float sCombSum[BM];

    const int bm = blockIdx.y * BM;
    const int bn = blockIdx.x * BN;
    const int t     = threadIdx.x;
    const int lane  = t & 31;
    const int wid   = t >> 5;
    const int waveM = wid >> 2;   // 0..1 -> 64 rows
    const int waveN = wid & 3;    // 0..3 -> 32 cols
    const int half  = lane >> 4;
    const int lm    = lane & 15;

    for (int i = t; i < BM * N_EXP; i += 256)
        sComb[i] = comb[(size_t)(bm + (i >> 3)) * N_EXP + (i & 7)];
    for (int i = t; i < BM; i += 256)
        sCombSum[i] = combsum[bm + i];

    const float ss = shared_scale[0];
    const float rs = router_scale[0];

    v8f outAcc[4][2];
#pragma unroll
    for (int ti = 0; ti < 4; ++ti)
#pragma unroll
        for (int tj = 0; tj < 2; ++tj) outAcc[ti][tj] = zero8();

    for (int e = 0; e < 9; ++e) {
        const float* Wf   = (e == 0) ? shared_w
                                     : expert_w + (size_t)(e - 1) * HIDDEN * HIDDEN;
        const unsigned short* Wb = wt + (size_t)e * HIDDEN * HIDDEN;  // [n][k]
        const float* bias = (e == 0) ? shared_b
                                     : expert_b + (size_t)(e - 1) * HIDDEN;

        v8f acc[4][2];
#pragma unroll
        for (int ti = 0; ti < 4; ++ti)
#pragma unroll
            for (int tj = 0; tj < 2; ++tj) acc[ti][tj] = zero8();

        for (int k0 = 0; k0 < HIDDEN; k0 += BK) {
            __syncthreads();
            if constexpr (PRE) {
                // pure bf16 copy staging: global b128 -> LDS b128, no VALU
#pragma unroll
                for (int i = 0; i < 2; ++i) {          // A: 512 chunks of 8 bf16
                    int idx = t + i * 256;
                    int m  = idx >> 2;
                    int k8 = (idx & 3) << 3;
                    *(v4u*)&sA[m * LDS_STRIDE + k8] =
                        *(const v4u*)&xbf[(size_t)(bm + m) * HIDDEN + k0 + k8];
                }
#pragma unroll
                for (int i = 0; i < 2; ++i) {          // B: 512 chunks of 8 bf16
                    int idx = t + i * 256;
                    int n  = idx >> 2;
                    int k8 = (idx & 3) << 3;
                    *(v4u*)&sB[n * LDS_STRIDE + k8] =
                        *(const v4u*)&Wb[(size_t)(bn + n) * HIDDEN + k0 + k8];
                }
                if (k0 + BK < HIDDEN)
                    __builtin_prefetch(&Wb[(size_t)(bn + (t >> 1)) * HIDDEN + k0 + BK + ((t & 1) << 4)], 0, 3);
            } else {
                // fallback: convert f32 -> bf16 while staging
#pragma unroll
                for (int i = 0; i < 4; ++i) {
                    int idx = t + i * 256;
                    int r  = idx >> 3;
                    int c4 = (idx & 7) << 2;
                    float4 f = *(const float4*)&x[(size_t)(bm + r) * HIDDEN + k0 + c4];
                    v2u p; p.x = pack_bf2(f.x, f.y); p.y = pack_bf2(f.z, f.w);
                    *(v2u*)&sA[r * LDS_STRIDE + c4] = p;
                }
#pragma unroll
                for (int i = 0; i < 2; ++i) {
                    int idx = t + i * 256;
                    int kp = idx >> 5;
                    int n4 = (idx & 31) << 2;
                    int k  = kp << 1;
                    float4 fa = *(const float4*)&Wf[(size_t)(k0 + k)     * HIDDEN + bn + n4];
                    float4 fb = *(const float4*)&Wf[(size_t)(k0 + k + 1) * HIDDEN + bn + n4];
                    *(unsigned int*)&sB[(n4 + 0) * LDS_STRIDE + k] = pack_bf2(fa.x, fb.x);
                    *(unsigned int*)&sB[(n4 + 1) * LDS_STRIDE + k] = pack_bf2(fa.y, fb.y);
                    *(unsigned int*)&sB[(n4 + 2) * LDS_STRIDE + k] = pack_bf2(fa.z, fb.z);
                    *(unsigned int*)&sB[(n4 + 3) * LDS_STRIDE + k] = pack_bf2(fa.w, fb.w);
                }
                if (k0 + BK < HIDDEN)
                    __builtin_prefetch(&Wf[(size_t)(k0 + BK + (t >> 3)) * HIDDEN + bn + ((t & 7) << 4)], 0, 3);
            }
            __syncthreads();

            // fragments: regs 0..3 hold K-pairs 0..7 = 16 contiguous bytes -> ds_load_b128 x2
            FragBF a[4], b[2];
#pragma unroll
            for (int ti = 0; ti < 4; ++ti) {
                int base = (waveM * 64 + ti * 16 + lm) * LDS_STRIDE + (half << 3);
                a[ti].q[0] = *(const v4u*)&sA[base];
                a[ti].q[1] = *(const v4u*)&sA[base + 16];
            }
#pragma unroll
            for (int tj = 0; tj < 2; ++tj) {
                int base = (waveN * 32 + tj * 16 + lm) * LDS_STRIDE + (half << 3);
                b[tj].q[0] = *(const v4u*)&sB[base];
                b[tj].q[1] = *(const v4u*)&sB[base + 16];
            }
#pragma unroll
            for (int ti = 0; ti < 4; ++ti)
#pragma unroll
                for (int tj = 0; tj < 2; ++tj)
                    acc[ti][tj] = __builtin_amdgcn_wmma_f32_16x16x32_bf16(
                        false, a[ti].v, false, b[tj].v,
                        (short)0, acc[ti][tj], false, false);
        }

        // fused epilogue: sigmoid + scale + accumulate
#pragma unroll
        for (int ti = 0; ti < 4; ++ti) {
#pragma unroll
            for (int tj = 0; tj < 2; ++tj) {
                int ng = bn + waveN * 32 + tj * 16 + lm;
                float bv = bias[ng];
#pragma unroll
                for (int v = 0; v < 8; ++v) {
                    int ml = waveM * 64 + ti * 16 + v + (half << 3);
                    float s = sigmoidf(acc[ti][tj][v] + bv);
                    float w = (e == 0) ? ss : rs * sComb[ml * N_EXP + (e - 1)];
                    outAcc[ti][tj][v] += w * s;
                }
            }
        }
    }

    // final: out = outAcc + x * (1 + rs * combsum)
#pragma unroll
    for (int ti = 0; ti < 4; ++ti) {
#pragma unroll
        for (int tj = 0; tj < 2; ++tj) {
            int ng = bn + waveN * 32 + tj * 16 + lm;
#pragma unroll
            for (int v = 0; v < 8; ++v) {
                int ml = waveM * 64 + ti * 16 + v + (half << 3);
                int mg = bm + ml;
                float xv = x[(size_t)mg * HIDDEN + ng];
                out[(size_t)mg * HIDDEN + ng] =
                    outAcc[ti][tj][v] + xv * (1.0f + rs * sCombSum[ml]);
            }
        }
    }
}

// ---------------------------------------------------------------- gate loss
__global__ void moe_loss_kernel(const float* __restrict__ gateSum,
                                float* __restrict__ out) {
    if (threadIdx.x == 0 && blockIdx.x == 0) {
        float loss = 0.0f;
#pragma unroll
        for (int e = 0; e < N_EXP; ++e) {
            float p = gateSum[e] / (float)BTOK;
            float d = (1.0f / (float)N_EXP) - p;
            loss += d * d;
        }
        loss = (loss / (float)N_EXP) * BAL_COEF;
        out[(size_t)BTOK * HIDDEN] = loss;
    }
}

// ---------------------------------------------------------------- launcher
extern "C" void kernel_launch(void* const* d_in, const int* in_sizes, int n_in,
                              void* d_out, int out_size, void* d_ws, size_t ws_size,
                              hipStream_t stream) {
    const float* x            = (const float*)d_in[0];
    const float* noise        = (const float*)d_in[1];
    const float* gate_w1      = (const float*)d_in[2];
    const float* gate_b1      = (const float*)d_in[3];
    const float* gate_w2      = (const float*)d_in[4];
    const float* gate_b2      = (const float*)d_in[5];
    const float* shared_w     = (const float*)d_in[6];
    const float* shared_b     = (const float*)d_in[7];
    const float* shared_scale = (const float*)d_in[8];
    const float* expert_w     = (const float*)d_in[9];
    const float* expert_b     = (const float*)d_in[10];
    const float* router_scale = (const float*)d_in[11];
    float* out = (float*)d_out;

    char* ws = (char*)d_ws;
    float* comb    = (float*)(ws);                                   // B*8 f32
    float* combsum = (float*)(ws + (size_t)BTOK * N_EXP * 4);        // B f32
    float* gateSum = (float*)(ws + (size_t)BTOK * N_EXP * 4 + (size_t)BTOK * 4);

    const size_t BF_OFF   = 256 * 1024;                              // 256KB aligned region
    const size_t XBF_SZ   = (size_t)BTOK * HIDDEN * 2;               // 16 MB
    const size_t WT_SZ    = (size_t)9 * HIDDEN * HIDDEN * 2;         // 75.5 MB
    const size_t NEED     = BF_OFF + XBF_SZ + WT_SZ;
    unsigned short* xbf = (unsigned short*)(ws + BF_OFF);
    unsigned short* wtb = (unsigned short*)(ws + BF_OFF + XBF_SZ);

    moe_init_kernel<<<1, 32, 0, stream>>>(gateSum);
    moe_gating_kernel<<<BTOK, 64, 0, stream>>>(x, noise, gate_w1, gate_b1,
                                               gate_w2, gate_b2,
                                               comb, combsum, gateSum);

    dim3 grid(HIDDEN / BN, BTOK / BM);
    if (ws_size >= NEED) {
        moe_cvt_x_kernel<<<(BTOK * HIDDEN) / (256 * 8), 256, 0, stream>>>(x, xbf);
        dim3 tgrid(HIDDEN / 64, HIDDEN / 64, 9);
        moe_cvt_w_kernel<<<tgrid, 256, 0, stream>>>(shared_w, expert_w, wtb);
        moe_gemm_kernel<true><<<grid, 256, 0, stream>>>(
            x, xbf, wtb, shared_w, shared_b, shared_scale,
            expert_w, expert_b, router_scale, comb, combsum, out);
    } else {
        moe_gemm_kernel<false><<<grid, 256, 0, stream>>>(
            x, xbf, wtb, shared_w, shared_b, shared_scale,
            expert_w, expert_b, router_scale, comb, combsum, out);
    }

    moe_loss_kernel<<<1, 32, 0, stream>>>(gateSum, out);
}
